// CausalSelfAttention_30107720745073
// MI455X (gfx1250) — matrix operand order repaired and checked by reading, not compiled, not run on hardware
//
#include <hip/hip_runtime.h>
#include <hip/hip_bf16.h>
#include <stdint.h>

typedef __attribute__((ext_vector_type(16))) _Float16 v16h;
typedef __attribute__((ext_vector_type(8)))  _Float16 v8h;
typedef __attribute__((ext_vector_type(8)))  float    v8f;
typedef __attribute__((ext_vector_type(4)))  unsigned int v4u;
typedef __attribute__((ext_vector_type(8)))  int      v8i_t;
typedef __attribute__((ext_vector_type(4)))  int      v4i_t;

#define B_  2
#define S_  2048
#define D_  1024
#define H_  16
#define HD_ 64

union V16U { v16h v; v8h h[2]; };

static __device__ __forceinline__ v8f wmma_f16(v16h a, v16h b, v8f c) {
  return __builtin_amdgcn_wmma_f32_16x16x32_f16(false, a, false, b, (short)0, c,
                                                false, false);
}

#if defined(__has_builtin)
#if __has_builtin(__builtin_amdgcn_tensor_load_to_lds)
#define HAVE_TDM 1
#endif
#endif

#ifdef HAVE_TDM
// Issue a TDM 2D tile load: tile_d0_u8 = row length in 8-byte units,
// tile_d1 = rows, stride_u8 = global row stride in 8-byte units.
// D# layout per CDNA5 ISA; tensor_dim == tile_dim
// so no OOB clipping is exercised. One call per wave = one TENSOR op.
static __device__ __forceinline__ void tdm_load_tile_2d(
    uint32_t lds_addr, const void* gptr, uint32_t tile_d0_u8, uint32_t tile_d1,
    uint32_t stride_u8) {
  uint64_t ga = (uint64_t)(uintptr_t)gptr;
  v4u g0;
  g0[0] = 1u;                                   // count=1 (valid), user mode
  g0[1] = lds_addr;                             // LDS byte address
  g0[2] = (uint32_t)ga;                         // global_addr[31:0]
  g0[3] = (uint32_t)((ga >> 32) & 0x01FFFFFFu)  // global_addr[56:32]
          | (2u << 30);                         // type = 2 ("image")
  v8i_t g1;
  g1[0] = (int)(3u << 16);                      // wg_mask=0, data_size=3 (8B)
  g1[1] = (int)((tile_d0_u8 & 0xFFFFu) << 16);  // tensor_dim0[15:0] @ bit48
  g1[2] = (int)((tile_d0_u8 >> 16) | ((tile_d1 & 0xFFFFu) << 16));
  g1[3] = (int)((tile_d1 >> 16) | (tile_d0_u8 << 16));  // tile_dim0 @ bit112
  g1[4] = (int)(tile_d1 & 0xFFFFu);             // tile_dim1; tile_dim2=0
  g1[5] = (int)stride_u8;                       // tensor_dim0_stride[31:0]
  g1[6] = 0;                                    // stride hi; dim1_stride=0
  g1[7] = 0;
  v4i_t z4 = {0, 0, 0, 0};
#if __clang_major__ >= 23
  v8i_t z8 = {0, 0, 0, 0, 0, 0, 0, 0};
  __builtin_amdgcn_tensor_load_to_lds(g0, g1, z4, z4, z8, 0);
#else
  __builtin_amdgcn_tensor_load_to_lds(g0, g1, z4, z4, 0);
#endif
}
#endif

// ---------------------------------------------------------------------------
// f32 -> f16 elementwise convert
// ---------------------------------------------------------------------------
__global__ void f32_to_f16_kernel(const float* __restrict__ src,
                                  _Float16* __restrict__ dst, int n) {
  int i = blockIdx.x * blockDim.x + threadIdx.x;
  int stride = gridDim.x * blockDim.x;
  for (; i < n; i += stride) dst[i] = (_Float16)src[i];
}

// ---------------------------------------------------------------------------
// C[M,N] = A[M,K] * W[N,K]^T + bias   (M=4096, N=K=1024)
// mode 0: f16 -> [b,h,s,hd]   (Q, K)
// mode 1: f16 -> [b,h,hd,s]   (V transposed for PV B-fragments)
// mode 2: f32 -> [b,s,D]      (final output projection)
// BM=BN=128, BK=64, 256 threads = 8 waves, each wave 32x64 (2x4 WMMA tiles).
// Tiles staged by the Tensor Data Mover, double-buffered: TDM fills buf^1
// while the 8 waves run WMMAs on buf. One barrier per K-step.
// ---------------------------------------------------------------------------
__global__ __launch_bounds__(256) void gemm_xwt_kernel(
    const _Float16* __restrict__ A, const _Float16* __restrict__ W,
    const float* __restrict__ bias, void* __restrict__ dst, int mode) {
  constexpr int K = 1024;
  constexpr int BM = 128, BK = 64;
  constexpr int NKB = K / BK;
  __shared__ _Float16 As[2][BM * BK];
  __shared__ _Float16 Bs[2][BM * BK];

  const int tid = threadIdx.x;
  const int lane = tid & 31;
  const int wave = tid >> 5;
  const int lane_n = lane & 15;
  const int halfsel = lane >> 4;  // 0: lanes 0-15, 1: lanes 16-31
  const int wm = wave & 3;        // 32-row group within 128
  const int wn = wave >> 2;       // 64-col group within 128
  const int m0 = blockIdx.y * BM;
  const int n0 = blockIdx.x * BM;

  v8f acc[2][4] = {};

#ifdef HAVE_TDM
  const uint32_t asb[2] = {(uint32_t)(uintptr_t)&As[0][0],
                           (uint32_t)(uintptr_t)&As[1][0]};
  const uint32_t bsb[2] = {(uint32_t)(uintptr_t)&Bs[0][0],
                           (uint32_t)(uintptr_t)&Bs[1][0]};
  if (wave == 0) {  // one wave issues the DMA (TDM ignores EXEC, per-wave op)
    tdm_load_tile_2d(asb[0], A + (size_t)m0 * K, BK / 4, BM, K / 4);
    tdm_load_tile_2d(bsb[0], W + (size_t)n0 * K, BK / 4, BM, K / 4);
  }
  for (int ib = 0; ib < NKB; ++ib) {
    const int buf = ib & 1;
    if (wave == 0) __builtin_amdgcn_s_wait_tensorcnt((short)0);
    __syncthreads();  // buf ready for all; buf^1 free (consumed last iter)
    if (wave == 0 && (ib + 1) < NKB) {
      const int kb2 = (ib + 1) * BK;
      tdm_load_tile_2d(asb[buf ^ 1], A + (size_t)m0 * K + kb2, BK / 4, BM,
                       K / 4);
      tdm_load_tile_2d(bsb[buf ^ 1], W + (size_t)n0 * K + kb2, BK / 4, BM,
                       K / 4);
    }
    const _Float16* Asb = &As[buf][0];
    const _Float16* Bsb = &Bs[buf][0];
#else
  for (int ib = 0; ib < NKB; ++ib) {
    const int kb = ib * BK;
#pragma unroll
    for (int i = 0; i < 4; ++i) {
      int chunk = tid + i * 256;
      int row = chunk >> 3;
      int c8 = (chunk & 7) * 8;
      *(v8h*)&As[0][row * BK + c8] =
          *(const v8h*)(A + (size_t)(m0 + row) * K + kb + c8);
      *(v8h*)&Bs[0][row * BK + c8] =
          *(const v8h*)(W + (size_t)(n0 + row) * K + kb + c8);
    }
    __syncthreads();
    const _Float16* Asb = &As[0][0];
    const _Float16* Bsb = &Bs[0][0];
#endif
#pragma unroll
    for (int kk = 0; kk < BK; kk += 32) {
      // A fragments (contiguous k order, same as B): row = lane%16,
      // two 8-half chunks at k-offsets (half?16:0) and +8.
      v16h af[2];
#pragma unroll
      for (int tr = 0; tr < 2; ++tr) {
        int row = wm * 32 + tr * 16 + lane_n;
        int c0 = kk + halfsel * 16;
        V16U u;
        u.h[0] = *(const v8h*)&Asb[row * BK + c0];
        u.h[1] = *(const v8h*)&Asb[row * BK + c0 + 8];
        af[tr] = u.v;
      }
      // B fragments (32x16): col n = lane%16, 16 contiguous halves at
      // k-offset (half?16:0).
      v16h bf[4];
#pragma unroll
      for (int tc = 0; tc < 4; ++tc) {
        int nrow = wn * 64 + tc * 16 + lane_n;
        bf[tc] = *(const v16h*)&Bsb[nrow * BK + kk + halfsel * 16];
      }
#pragma unroll
      for (int tr = 0; tr < 2; ++tr)
#pragma unroll
        for (int tc = 0; tc < 4; ++tc)
          acc[tr][tc] = wmma_f16(af[tr], bf[tc], acc[tr][tc]);
    }
#ifndef HAVE_TDM
    __syncthreads();
#endif
  }

  float bcol[4];
#pragma unroll
  for (int tc = 0; tc < 4; ++tc)
    bcol[tc] = bias[n0 + wn * 64 + tc * 16 + lane_n];

  // C/D layout: vgpr r, lane L -> row r + (L<16?0:8), col L%16.
#pragma unroll
  for (int tr = 0; tr < 2; ++tr) {
#pragma unroll
    for (int tc = 0; tc < 4; ++tc) {
#pragma unroll
      for (int r = 0; r < 8; ++r) {
        int t = m0 + wm * 32 + tr * 16 + r + halfsel * 8;  // token (b*S+s)
        int f = n0 + wn * 64 + tc * 16 + lane_n;           // feature
        float val = acc[tr][tc][r] + bcol[tc];
        int b = t >> 11;
        int s = t & (S_ - 1);
        int h = f >> 6;
        int hd = f & (HD_ - 1);
        if (mode == 0) {
          ((_Float16*)dst)[(((size_t)b * H_ + h) * S_ + s) * HD_ + hd] =
              (_Float16)val;
        } else if (mode == 1) {
          ((_Float16*)dst)[(((size_t)b * H_ + h) * HD_ + hd) * S_ + s] =
              (_Float16)val;
        } else {
          ((float*)dst)[(size_t)t * D_ + f] = val;
        }
      }
    }
  }
}

// ---------------------------------------------------------------------------
// Causal flash attention. One wave owns one 16-row Q tile of one (b,h).
// Q,K: [b,h,s,hd] f16; Vt: [b,h,hd,s] f16; out: [b,s,D] f16 (pre-Wo).
// ---------------------------------------------------------------------------
__global__ __launch_bounds__(128) void attn_kernel(
    const _Float16* __restrict__ Q, const _Float16* __restrict__ Km,
    const _Float16* __restrict__ Vt, _Float16* __restrict__ O) {
  __shared__ _Float16 pshm[4][16 * 32];  // per-wave P layout-shuffle scratch
  const int lane = threadIdx.x & 31;
  const int wave = threadIdx.x >> 5;
  const int lane_n = lane & 15;
  const int halfsel = lane >> 4;
  const int bh = blockIdx.x;             // b*H + h
  const int qt = blockIdx.y * 4 + wave;  // 0..127
  const int q0 = qt * 16;
  const int b = bh >> 4;
  const int h = bh & (H_ - 1);

  const _Float16* Qb = Q + ((size_t)bh * S_ + q0) * HD_;
  const _Float16* Kb = Km + (size_t)bh * S_ * HD_;
  const _Float16* Vb = Vt + (size_t)bh * HD_ * S_;

  // Q fragments for d=[0,32) and [32,64), pre-scaled by 1/sqrt(64)=0.125.
  v16h qf[2];
#pragma unroll
  for (int kk = 0; kk < 2; ++kk) {
    const _Float16* p = Qb + lane_n * HD_ + kk * 32 + halfsel * 16;
    V16U u;
    u.h[0] = *(const v8h*)p;
    u.h[1] = *(const v8h*)(p + 8);
    qf[kk] = u.v * (_Float16)0.125f;
  }

  float m_r[8], l_r[8];
  v8f o_acc[4] = {};
#pragma unroll
  for (int r = 0; r < 8; ++r) { m_r[r] = -1e30f; l_r[r] = 0.0f; }

  const int jmax = q0 + 16;
  for (int j0 = 0; j0 < jmax; j0 += 32) {  // uniform per wave: EXEC stays full
    if (j0 + 32 < jmax) {                  // hint next K/V tiles into caches
      __builtin_prefetch(Kb + (size_t)(j0 + 32 + lane_n) * HD_, 0, 1);
      __builtin_prefetch(Vb + (size_t)lane_n * S_ + j0 + 32, 0, 1);
    }
    // K^T fragments: B[k=d][n=j]: lane holds key row j0+t*16+lane%16,
    // 16 contiguous d-values at offset kk*32 + (half?16:0).
    v16h kf[2][2];
#pragma unroll
    for (int t = 0; t < 2; ++t)
#pragma unroll
      for (int kk = 0; kk < 2; ++kk)
        kf[t][kk] = *(const v16h*)(Kb + (size_t)(j0 + t * 16 + lane_n) * HD_ +
                                   kk * 32 + halfsel * 16);

    v8f s0 = {}, s1 = {};
    s0 = wmma_f16(qf[0], kf[0][0], s0);
    s0 = wmma_f16(qf[1], kf[0][1], s0);
    s1 = wmma_f16(qf[0], kf[1][0], s1);
    s1 = wmma_f16(qf[1], kf[1][1], s1);

    // Online softmax: row r+halfsel*8 lives in this half's 16 lanes.
    float p0[8], p1[8], alpha[8];
#pragma unroll
    for (int r = 0; r < 8; ++r) {
      int rowq = q0 + r + halfsel * 8;
      float a = s0[r], c = s1[r];
      if (j0 + lane_n > rowq) a = -1e30f;
      if (j0 + 16 + lane_n > rowq) c = -1e30f;
      float mx = fmaxf(a, c);
#pragma unroll
      for (int msk = 8; msk >= 1; msk >>= 1)
        mx = fmaxf(mx, __shfl_xor(mx, msk, 32));
      float mnew = fmaxf(m_r[r], mx);
      alpha[r] = __expf(m_r[r] - mnew);
      float e0 = __expf(a - mnew);
      float e1 = __expf(c - mnew);
      float rs = e0 + e1;
#pragma unroll
      for (int msk = 8; msk >= 1; msk >>= 1)
        rs += __shfl_xor(rs, msk, 32);
      l_r[r] = l_r[r] * alpha[r] + rs;
      m_r[r] = mnew;
      p0[r] = e0;
      p1[r] = e1;
    }
#pragma unroll
    for (int dt = 0; dt < 4; ++dt)
#pragma unroll
      for (int r = 0; r < 8; ++r) o_acc[dt][r] *= alpha[r];

    // C-layout P -> A-fragment layout via per-wave LDS (LDS is in-order;
    // the wait + memory clobber stops compiler reordering across lanes).
#pragma unroll
    for (int r = 0; r < 8; ++r) {
      int mrow = r + halfsel * 8;
      pshm[wave][mrow * 32 + lane_n] = (_Float16)p0[r];
      pshm[wave][mrow * 32 + 16 + lane_n] = (_Float16)p1[r];
    }
    asm volatile("s_wait_dscnt 0x0" ::: "memory");
    V16U pu;
    pu.h[0] = *(const v8h*)&pshm[wave][lane_n * 32 + halfsel * 16];
    pu.h[1] = *(const v8h*)&pshm[wave][lane_n * 32 + halfsel * 16 + 8];
    v16h pf = pu.v;

    // O += P * V using transposed V: 32 contiguous bytes per lane.
#pragma unroll
    for (int dt = 0; dt < 4; ++dt) {
      v16h vf = *(const v16h*)(Vb + (size_t)(dt * 16 + lane_n) * S_ + j0 +
                               halfsel * 16);
      o_acc[dt] = wmma_f16(pf, vf, o_acc[dt]);
    }
  }

  // Normalize, write [b, s, h*64+d] as f16 for the Wo GEMM.
#pragma unroll
  for (int dt = 0; dt < 4; ++dt)
#pragma unroll
    for (int r = 0; r < 8; ++r) {
      int srow = q0 + r + halfsel * 8;
      int f = h * HD_ + dt * 16 + lane_n;
      O[((size_t)b * S_ + srow) * D_ + f] = (_Float16)(o_acc[dt][r] / l_r[r]);
    }
}

// ---------------------------------------------------------------------------
extern "C" void kernel_launch(void* const* d_in, const int* in_sizes, int n_in,
                              void* d_out, int out_size, void* d_ws,
                              size_t ws_size, hipStream_t stream) {
  const float* x = (const float*)d_in[0];
  const float* Wq = (const float*)d_in[1];
  const float* bq = (const float*)d_in[2];
  const float* Wk = (const float*)d_in[3];
  const float* bk = (const float*)d_in[4];
  const float* Wv = (const float*)d_in[5];
  const float* bv = (const float*)d_in[6];
  const float* Wo = (const float*)d_in[7];
  const float* bo = (const float*)d_in[8];

  char* ws = (char*)d_ws;
  const size_t MB = 1ull << 20;
  _Float16* x16 = (_Float16*)(ws + 0);        // 8 MB  [B*S, D]
  _Float16* wq16 = (_Float16*)(ws + 8 * MB);  // 2 MB each
  _Float16* wk16 = (_Float16*)(ws + 10 * MB);
  _Float16* wv16 = (_Float16*)(ws + 12 * MB);
  _Float16* wo16 = (_Float16*)(ws + 14 * MB);
  _Float16* Q16 = (_Float16*)(ws + 16 * MB);   // 8 MB [b,h,s,hd]
  _Float16* K16 = (_Float16*)(ws + 24 * MB);   // 8 MB [b,h,s,hd]
  _Float16* Vt16 = (_Float16*)(ws + 32 * MB);  // 8 MB [b,h,hd,s]
  _Float16* A16 = (_Float16*)(ws + 40 * MB);   // 8 MB [b,s,D]

  const int NX = B_ * S_ * D_;
  const int NW = D_ * D_;
  f32_to_f16_kernel<<<1024, 256, 0, stream>>>(x, x16, NX);
  f32_to_f16_kernel<<<512, 256, 0, stream>>>(Wq, wq16, NW);
  f32_to_f16_kernel<<<512, 256, 0, stream>>>(Wk, wk16, NW);
  f32_to_f16_kernel<<<512, 256, 0, stream>>>(Wv, wv16, NW);
  f32_to_f16_kernel<<<512, 256, 0, stream>>>(Wo, wo16, NW);

  dim3 gg(D_ / 128, (B_ * S_) / 128);  // (8, 32)
  gemm_xwt_kernel<<<gg, 256, 0, stream>>>(x16, wq16, bq, (void*)Q16, 0);
  gemm_xwt_kernel<<<gg, 256, 0, stream>>>(x16, wk16, bk, (void*)K16, 0);
  gemm_xwt_kernel<<<gg, 256, 0, stream>>>(x16, wv16, bv, (void*)Vt16, 1);

  attn_kernel<<<dim3(B_ * H_, S_ / 64), 128, 0, stream>>>(Q16, K16, Vt16, A16);

  gemm_xwt_kernel<<<gg, 256, 0, stream>>>(A16, wo16, bo, d_out, 2);
}
